// MoleculeGenerator_56573309223978
// MI455X (gfx1250) — matrix operand order repaired and checked by reading, not compiled, not run on hardware
//
#include <hip/hip_runtime.h>
#include <hip/hip_bf16.h>

typedef _Float16 v16h __attribute__((ext_vector_type(16)));
typedef float    v8f  __attribute__((ext_vector_type(8)));

#define N_ROWS    200000
#define ROW_TILES 12500      // N_ROWS / 16
#define A_DIM     128
#define H_DIM     1024
#define RSPLIT    32         // row-split blocks per column tile
#define NB_DEC    128        // persistent decoder blocks (must all be resident)

__device__ __forceinline__ float waveRed(float v) {
  #pragma unroll
  for (int o = 16; o > 0; o >>= 1) v += __shfl_down(v, o, 32);
  return v;
}

__device__ __forceinline__ void cvt4(v16h& v, int base, float4 f) {
  v[base + 0] = (_Float16)f.x;
  v[base + 1] = (_Float16)f.y;
  v[base + 2] = (_Float16)f.z;
  v[base + 3] = (_Float16)f.w;
}

// Device-wide split barrier: monotonic counter, target = barrier_index * NB_DEC.
__device__ __forceinline__ void grid_sync(unsigned int* bar, unsigned int target) {
  __threadfence();                 // flush prior global writes to device scope
  __syncthreads();
  if (threadIdx.x == 0) {
    __hip_atomic_fetch_add(bar, 1u, __ATOMIC_RELEASE, __HIP_MEMORY_SCOPE_AGENT);
    while (__hip_atomic_load(bar, __ATOMIC_ACQUIRE, __HIP_MEMORY_SCOPE_AGENT) < target)
      __builtin_amdgcn_s_sleep(1);
  }
  __syncthreads();
}

// ---- zero accumulator + barrier counter ----
__global__ void k_zero(float* __restrict__ p, int n) {
  int i = blockIdx.x * blockDim.x + threadIdx.x;
  if (i < n) p[i] = 0.f;
}

// ---- encoder layer 1: acc[h] += sum_n relu(x[n,:] @ W1[h,:] + b1[h]) via WMMA ----
__global__ __launch_bounds__(256) void k_enc(const float* __restrict__ x,
                                             const float* __restrict__ W1,
                                             const float* __restrict__ b1,
                                             float* __restrict__ acc) {
  const int ct   = blockIdx.x & 63;        // column tile (H / 16)
  const int rs   = blockIdx.x >> 6;        // row-split index 0..RSPLIT-1
  const int wave = threadIdx.x >> 5;
  const int lane = threadIdx.x & 31;

  const int m     = lane & 15;                  // A row within tile
  const int abase = (lane < 16) ? 0 : 8;        // A K-phase per ISA layout
  const int kbB   = (lane < 16) ? 0 : 8;        // B K-phase per ISA layout
  const int n     = lane & 15;                  // B column within tile
  const int h0    = ct * 16;

  // Preload + convert the B (weight) fragments once; reused for every row tile.
  const float* wrow = W1 + (size_t)(h0 + n) * A_DIM;
  v16h btile[4];
  #pragma unroll
  for (int kc = 0; kc < 4; ++kc) {
    const float4* bp4 = reinterpret_cast<const float4*>(wrow + kc * 32 + kbB);
    cvt4(btile[kc], 0,  bp4[0]);
    cvt4(btile[kc], 4,  bp4[1]);
    cvt4(btile[kc], 8,  bp4[4]);
    cvt4(btile[kc], 12, bp4[5]);
  }
  const float b1v = b1[h0 + n];

  float colpart = 0.f;
  const int wv = rs * 8 + wave;                 // 0..255 workers per column tile
  for (int rt = wv; rt < ROW_TILES; rt += RSPLIT * 8) {
    const float* xrow = x + (size_t)(rt * 16 + m) * A_DIM;
    v8f d = {};
    #pragma unroll
    for (int kc = 0; kc < 4; ++kc) {
      const float4* ap4 = reinterpret_cast<const float4*>(xrow + kc * 32 + abase);
      v16h a;
      cvt4(a, 0,  ap4[0]);   // K = base+0..3
      cvt4(a, 4,  ap4[1]);   // K = base+4..7
      cvt4(a, 8,  ap4[4]);   // K = base+16..19
      cvt4(a, 12, ap4[5]);   // K = base+20..23
      d = __builtin_amdgcn_wmma_f32_16x16x32_f16(false, a, false, btile[kc],
                                                 (short)0, d, false, false);
    }
    #pragma unroll
    for (int r = 0; r < 8; ++r) {
      float v = d[r] + b1v;
      colpart += (v > 0.f) ? v : 0.f;
    }
  }
  // lanes l and l^16 hold the two row-halves of the same output column
  float other = __shfl_xor(colpart, 16, 32);
  if (lane < 16) atomicAdd(&acc[h0 + lane], colpart + other);
}

// ---- h = (acc/N) @ W2^T + b2 ----
__global__ __launch_bounds__(256) void k_h(const float* __restrict__ acc,
                                           const float* __restrict__ W2,
                                           const float* __restrict__ b2,
                                           float* __restrict__ h) {
  const int i    = blockIdx.x * 8 + (threadIdx.x >> 5);
  const int lane = threadIdx.x & 31;
  const float4* wr = (const float4*)(W2 + (size_t)i * H_DIM);
  const float4* a4 = (const float4*)acc;
  float s = 0.f;
  for (int j = lane; j < H_DIM / 4; j += 32) {
    float4 w = wr[j], a = a4[j];
    s += w.x * a.x + w.y * a.y + w.z * a.z + w.w * a.w;
  }
  s = waveRed(s);
  if (lane == 0) h[i] = s * (1.0f / (float)N_ROWS) + b2[i];
}

// ---- mu / logvar heads + reparameterize; init LSTM state ----
__global__ __launch_bounds__(256) void k_heads(const float* __restrict__ h,
                                               const float* __restrict__ muW,
                                               const float* __restrict__ mub,
                                               const float* __restrict__ lvW,
                                               const float* __restrict__ lvb,
                                               const float* __restrict__ eps,
                                               float* __restrict__ out_mu,
                                               float* __restrict__ out_lv,
                                               float* __restrict__ hT,
                                               float* __restrict__ cT) {
  const int i    = blockIdx.x * 8 + (threadIdx.x >> 5);
  const int lane = threadIdx.x & 31;
  const float4* mr = (const float4*)(muW + (size_t)i * H_DIM);
  const float4* lr = (const float4*)(lvW + (size_t)i * H_DIM);
  const float4* h4 = (const float4*)h;
  float sm = 0.f, sl = 0.f;
  for (int j = lane; j < H_DIM / 4; j += 32) {
    float4 hv = h4[j], a = mr[j], b = lr[j];
    sm += a.x * hv.x + a.y * hv.y + a.z * hv.z + a.w * hv.w;
    sl += b.x * hv.x + b.y * hv.y + b.z * hv.z + b.w * hv.w;
  }
  sm = waveRed(sm);
  sl = waveRed(sl);
  if (lane == 0) {
    float mu = sm + mub[i];
    float lv = sl + lvb[i];
    out_mu[i] = mu;
    out_lv[i] = lv;
    hT[i] = mu + eps[i] * expf(0.5f * lv);
    cT[i] = 0.f;
  }
}

// ---- persistent decoder: all 256 LSTM steps in one launch, grid-sync'd ----
__global__ __launch_bounds__(256) void k_decode(const float* __restrict__ Wih,
                                                const float* __restrict__ bih,
                                                const float* __restrict__ Whh,
                                                const float* __restrict__ bhh,
                                                const float* __restrict__ apW1,
                                                const float* __restrict__ apb1,
                                                const float* __restrict__ apW2,
                                                const float* __restrict__ apb2,
                                                float* __restrict__ hT,
                                                float* __restrict__ cT,
                                                float* __restrict__ gates,
                                                float* __restrict__ t1,
                                                unsigned int* __restrict__ bar,
                                                float* __restrict__ out_logits) {
  const int gwave = blockIdx.x * 8 + (threadIdx.x >> 5);  // 0..1023
  const int lane  = threadIdx.x & 31;
  const int tid   = blockIdx.x * 256 + threadIdx.x;
  unsigned int target = 0;

  for (int t = 0; t < 256; ++t) {
    // ---- phase 1: gates[r] = W_ih[r,:1024]·h + W_hh[r,:]·h + b_ih[r] + b_hh[r]
    for (int r = gwave; r < 4096; r += 1024) {
      const float4* wi = (const float4*)(Wih + (size_t)r * 1056);  // pad cols unused
      const float4* wh = (const float4*)(Whh + (size_t)r * H_DIM);
      const float4* h4 = (const float4*)hT;
      float s = 0.f;
      for (int j = lane; j < H_DIM / 4; j += 32) {
        float4 hv = h4[j], a = wi[j], b = wh[j];
        s += a.x * hv.x + a.y * hv.y + a.z * hv.z + a.w * hv.w;
        s += b.x * hv.x + b.y * hv.y + b.z * hv.z + b.w * hv.w;
      }
      s = waveRed(s);
      if (lane == 0) gates[r] = s + bih[r] + bhh[r];
    }
    target += NB_DEC; grid_sync(bar, target);

    // ---- phase 2: cell update (torch gate order i,f,g,o)
    if (tid < 1024) {
      float ig = gates[tid];
      float fg = gates[1024 + tid];
      float gg = gates[2048 + tid];
      float og = gates[3072 + tid];
      float si = 1.f / (1.f + expf(-ig));
      float sf = 1.f / (1.f + expf(-fg));
      float so = 1.f / (1.f + expf(-og));
      float c = sf * cT[tid] + si * tanhf(gg);
      cT[tid] = c;
      hT[tid] = so * tanhf(c);
    }
    target += NB_DEC; grid_sync(bar, target);

    // ---- phase 3: t1 = relu(ap_W1 @ h + ap_b1), one row per wave
    {
      const float4* w4 = (const float4*)(apW1 + (size_t)gwave * H_DIM);
      const float4* h4 = (const float4*)hT;
      float s = 0.f;
      for (int j = lane; j < H_DIM / 4; j += 32) {
        float4 a = w4[j], hv = h4[j];
        s += a.x * hv.x + a.y * hv.y + a.z * hv.z + a.w * hv.w;
      }
      s = waveRed(s);
      if (lane == 0) {
        float v = s + apb1[gwave];
        t1[gwave] = (v > 0.f) ? v : 0.f;
      }
    }
    target += NB_DEC; grid_sync(bar, target);

    // ---- phase 4: logits row = ap_W2 @ t1 + ap_b2 (128 rows)
    if (gwave < 128) {
      const float4* w4 = (const float4*)(apW2 + (size_t)gwave * H_DIM);
      const float4* t4 = (const float4*)t1;
      float s = 0.f;
      for (int j = lane; j < H_DIM / 4; j += 32) {
        float4 a = w4[j], tv = t4[j];
        s += a.x * tv.x + a.y * tv.y + a.z * tv.z + a.w * tv.w;
      }
      s = waveRed(s);
      if (lane == 0) out_logits[t * 128 + gwave] = s + apb2[gwave];
    }
    target += NB_DEC; grid_sync(bar, target);
  }
}

extern "C" void kernel_launch(void* const* d_in, const int* in_sizes, int n_in,
                              void* d_out, int out_size, void* d_ws, size_t ws_size,
                              hipStream_t stream) {
  const float* x      = (const float*)d_in[0];
  // d_in[1] = edge_index (int64) — unused by the reference output
  const float* eps    = (const float*)d_in[2];
  const float* enc_W1 = (const float*)d_in[3];
  const float* enc_b1 = (const float*)d_in[4];
  const float* enc_W2 = (const float*)d_in[5];
  const float* enc_b2 = (const float*)d_in[6];
  const float* mu_W   = (const float*)d_in[7];
  const float* mu_b   = (const float*)d_in[8];
  const float* lv_W   = (const float*)d_in[9];
  const float* lv_b   = (const float*)d_in[10];
  const float* W_ih   = (const float*)d_in[11];
  const float* b_ih   = (const float*)d_in[12];
  const float* W_hh   = (const float*)d_in[13];
  const float* b_hh   = (const float*)d_in[14];
  const float* ap_W1  = (const float*)d_in[15];
  const float* ap_b1  = (const float*)d_in[16];
  const float* ap_W2  = (const float*)d_in[17];
  const float* ap_b2  = (const float*)d_in[18];

  float* out        = (float*)d_out;
  float* out_logits = out;           // [256,128]
  float* out_mu     = out + 32768;   // [1024]
  float* out_lv     = out + 33792;   // [1024]

  float* ws    = (float*)d_ws;
  float* acc   = ws;                         // [1024] colsums of relu layer-1
  float* h     = ws + 1024;                  // [1024]
  float* hT    = ws + 2048;                  // [1024] LSTM hidden
  float* cT    = ws + 3072;                  // [1024] LSTM cell
  float* gates = ws + 4096;                  // [4096]
  float* t1    = ws + 8192;                  // [1024]
  unsigned int* bar = (unsigned int*)(ws + 9216);  // barrier counter

  // Zero acc + h..t1 region + barrier counter (harness poisons ws with 0xAA)
  k_zero<<<37, 256, 0, stream>>>(ws, 9220);

  // Encoder (WMMA); mean folded through the second (linear) encoder layer
  k_enc<<<64 * RSPLIT, 256, 0, stream>>>(x, enc_W1, enc_b1, acc);
  k_h<<<128, 256, 0, stream>>>(acc, enc_W2, enc_b2, h);
  k_heads<<<128, 256, 0, stream>>>(h, mu_W, mu_b, lv_W, lv_b, eps,
                                   out_mu, out_lv, hT, cT);

  // Persistent sequential decode: 256 steps, 4 device-wide barriers per step
  k_decode<<<NB_DEC, 256, 0, stream>>>(W_ih, b_ih, W_hh, b_hh,
                                       ap_W1, ap_b1, ap_W2, ap_b2,
                                       hT, cT, gates, t1, bar, out_logits);
}
